// SelfAttnMatch_31799937860228
// MI455X (gfx1250) — matrix operand order repaired and checked by reading, not compiled, not run on hardware
//
#include <hip/hip_runtime.h>

typedef __attribute__((ext_vector_type(16))) __bf16 v16bf;
typedef __attribute__((ext_vector_type(8)))  __bf16 v8bf;
typedef __attribute__((ext_vector_type(8)))  float  v8f;

#define B_ 8
#define L_ 2048
#define D_ 768

static __device__ __forceinline__ v16bf combine16(v8bf lo, v8bf hi) {
  v16bf r;
#pragma unroll
  for (int i = 0; i < 8; ++i) { r[i] = lo[i]; r[i + 8] = hi[i]; }
  return r;
}

// A operand (16x32 bf16). Row-major source, rows r0..r0+15, K window starting at kk.
// ISA layout: lanes 0-15 hold row (lane), K = {kk..kk+7, kk+16..kk+23};
//             lanes 16-31 hold row (lane-16), K = {kk+8..kk+15, kk+24..kk+31}.
static __device__ __forceinline__ v16bf load_a(const __bf16* base, int ld, int r0,
                                               int kk, int lane) {
  const __bf16* rowp = base + (size_t)(r0 + (lane & 15)) * ld + kk + ((lane >> 4) << 3);
  v8bf lo = *(const v8bf*)(rowp);        // 16B aligned
  v8bf hi = *(const v8bf*)(rowp + 16);   // 16B aligned
  return combine16(lo, hi);
}

// B operand (32x16 bf16) where B[k][n] = src[n0+n][kk+k] (transposed-B source,
// row-major rows of length ld). ISA layout: lanes 0-15 = col n, K={kk..kk+7, kk+16..kk+23};
// lanes 16-31 = col (lane-16), K={kk+8..kk+15, kk+24..kk+31} -> two 16B loads per lane.
static __device__ __forceinline__ v16bf load_bt(const __bf16* base, int ld, int n0,
                                                int kk, int lane) {
  const __bf16* p = base + (size_t)(n0 + (lane & 15)) * ld + kk + ((lane >> 4) << 3);
  v8bf lo = *(const v8bf*)(p);           // 16B aligned
  v8bf hi = *(const v8bf*)(p + 16);      // 16B aligned
  return combine16(lo, hi);
}

static __device__ __forceinline__ v8f wmma_bf16(v16bf a, v16bf b, v8f c) {
  return __builtin_amdgcn_wmma_f32_16x16x32_bf16(false, a, false, b, (short)0, c,
                                                 false, false);
}

// ---------------- fp32 -> bf16 elementwise convert ----------------
__global__ void sam_cvt_bf16_kernel(const float* __restrict__ src,
                                    __bf16* __restrict__ dst, int n) {
  int i = blockIdx.x * blockDim.x + threadIdx.x;
  int stride = gridDim.x * blockDim.x;
  for (; i < n; i += stride) dst[i] = (__bf16)src[i];
}

// ---------------- x[B,L,D] fp32 -> x_t[B,D,L] bf16 (LDS tiled transpose) -------
__global__ void sam_transpose_kernel(const float* __restrict__ x,
                                     __bf16* __restrict__ xt) {
  __shared__ float tile[16][17];
  int b = blockIdx.z;
  int d0 = blockIdx.x * 16, l0 = blockIdx.y * 16;
  int tx = threadIdx.x, ty = threadIdx.y;
  tile[ty][tx] = x[((size_t)b * L_ + (l0 + ty)) * D_ + d0 + tx];
  __syncthreads();
  xt[((size_t)b * D_ + (d0 + ty)) * L_ + l0 + tx] = (__bf16)tile[tx][ty];
}

// ---------------- x_proj = relu(x @ W^T + b), bf16 out ----------------
// one wave per 16x64 output tile (4 accumulators); each A load feeds 4 WMMAs
__global__ void sam_proj_kernel(const __bf16* __restrict__ xb,
                                const __bf16* __restrict__ wb,
                                const float* __restrict__ bias,
                                __bf16* __restrict__ xproj) {
  int lane = threadIdx.x;
  int n0 = blockIdx.x * 64;        // output feature chunk (4 tiles)
  int r0 = blockIdx.y * 16;        // row tile over B*L
  v8f c[4] = {};
  for (int kk = 0; kk < D_; kk += 32) {
    v16bf a = load_a(xb, D_, r0, kk, lane);
#pragma unroll
    for (int t = 0; t < 4; ++t) {
      v16bf bm = load_bt(wb, D_, n0 + t * 16, kk, lane);  // B[k][n] = W[n][k]
      c[t] = wmma_bf16(a, bm, c[t]);
    }
  }
  int half = lane >> 4;
#pragma unroll
  for (int t = 0; t < 4; ++t) {
    int col = n0 + t * 16 + (lane & 15);
    float bv = bias[col];
#pragma unroll
    for (int r = 0; r < 8; ++r) {
      float v = fmaxf(c[t][r] + bv, 0.f);
      xproj[(size_t)(r0 + r + 8 * half) * D_ + col] = (__bf16)v;
    }
  }
}

// ---------------- scores + softmax -> alpha (bf16) ----------------
// one wave per (batch, 16-row query tile). Q tile staged in LDS (24 KB).
// Key dim blocked 8 tiles (128 keys) per step: one A read feeds 8 WMMAs and
// the online-softmax reduction does a single shuffle tree per row per block.
__global__ void sam_softmax_kernel(const __bf16* __restrict__ xproj,
                                   __bf16* __restrict__ alpha) {
  __shared__ __bf16 sQ[16 * D_];   // 24 KB
  int lane = threadIdx.x;
  int b  = blockIdx.x >> 7;          // /128 query tiles
  int qt = blockIdx.x & 127;
  const __bf16* xp = xproj + (size_t)b * L_ * D_;
  int q0 = qt * 16;
  int half = lane >> 4;

  // cooperative coalesced load of the 16 contiguous Q rows into LDS
  {
    const __bf16* src = xp + (size_t)q0 * D_;
#pragma unroll 4
    for (int idx = lane; idx < (16 * D_) / 8; idx += 32)
      *(v8bf*)&sQ[idx * 8] = *(const v8bf*)(src + idx * 8);
  }
  __syncthreads();

  int arow = (lane & 15) * D_;                 // A-operand row base in LDS
  int koff = (lane >> 4) << 3;

  float m[8], l[8];
#pragma unroll
  for (int r = 0; r < 8; ++r) { m[r] = -3.4e38f; l[r] = 0.f; }

  for (int k0 = 0; k0 < L_; k0 += 128) {
    v8f c[8] = {};
    for (int kk = 0; kk < D_; kk += 32) {
      v8bf lo = *(const v8bf*)&sQ[arow + kk + koff];
      v8bf hi = *(const v8bf*)&sQ[arow + kk + koff + 16];
      v16bf a = combine16(lo, hi);
#pragma unroll
      for (int t = 0; t < 8; ++t) {
        v16bf bm = load_bt(xp, D_, k0 + t * 16, kk, lane);
        c[t] = wmma_bf16(a, bm, c[t]);
      }
    }
#pragma unroll
    for (int r = 0; r < 8; ++r) {
      // all 8 tiles share row (q0 + r + 8*half): reduce across tiles first
      float vmax = c[0][r];
#pragma unroll
      for (int t = 1; t < 8; ++t) vmax = fmaxf(vmax, c[t][r]);
#pragma unroll
      for (int off = 1; off < 16; off <<= 1)
        vmax = fmaxf(vmax, __shfl_xor(vmax, off, 32));
      float mn = fmaxf(m[r], vmax);
      float s = 0.f;
#pragma unroll
      for (int t = 0; t < 8; ++t) s += __expf(c[t][r] - mn);
#pragma unroll
      for (int off = 1; off < 16; off <<= 1)
        s += __shfl_xor(s, off, 32);
      l[r] = l[r] * __expf(m[r] - mn) + s;
      m[r] = mn;
    }
  }

  float linv[8];
#pragma unroll
  for (int r = 0; r < 8; ++r) linv[r] = 1.f / l[r];

  // pass 2: recompute S blocks and emit normalized alpha (bf16)
  __bf16* adst = alpha + (size_t)b * L_ * L_;
  for (int k0 = 0; k0 < L_; k0 += 128) {
    v8f c[8] = {};
    for (int kk = 0; kk < D_; kk += 32) {
      v8bf lo = *(const v8bf*)&sQ[arow + kk + koff];
      v8bf hi = *(const v8bf*)&sQ[arow + kk + koff + 16];
      v16bf a = combine16(lo, hi);
#pragma unroll
      for (int t = 0; t < 8; ++t) {
        v16bf bm = load_bt(xp, D_, k0 + t * 16, kk, lane);
        c[t] = wmma_bf16(a, bm, c[t]);
      }
    }
#pragma unroll
    for (int t = 0; t < 8; ++t) {
      int col = k0 + t * 16 + (lane & 15);
#pragma unroll
      for (int r = 0; r < 8; ++r) {
        float p = __expf(c[t][r] - m[r]) * linv[r];
        adst[(size_t)(q0 + r + 8 * half) * L_ + col] = (__bf16)p;
      }
    }
  }
}

// ---------------- out = alpha @ x (fp32 out) ----------------
// one wave per (batch, query tile, 128-wide D chunk): 8 accumulators (64 VGPRs)
__global__ void sam_pv_kernel(const __bf16* __restrict__ alpha,
                              const __bf16* __restrict__ xt,
                              float* __restrict__ out) {
  int lane = threadIdx.x;
  int dchunk = blockIdx.x;           // 0..5
  int qt = blockIdx.y;               // 0..127
  int b  = blockIdx.z;               // 0..7
  const __bf16* arow = alpha + (size_t)b * L_ * L_;
  const __bf16* xtb  = xt    + (size_t)b * D_ * L_;
  int q0 = qt * 16;
  int d0 = dchunk * 128;

  v8f o[8] = {};
  for (int k0 = 0; k0 < L_; k0 += 32) {
    v16bf a = load_a(arow, L_, q0, k0, lane);   // alpha rows, contiguous keys
#pragma unroll
    for (int t = 0; t < 8; ++t) {
      v16bf bm = load_bt(xtb, L_, d0 + t * 16, k0, lane);  // B[k][n] = x[k][n]
      o[t] = wmma_bf16(a, bm, o[t]);
    }
  }
  int half = lane >> 4;
#pragma unroll
  for (int t = 0; t < 8; ++t) {
    int col = d0 + t * 16 + (lane & 15);
#pragma unroll
    for (int r = 0; r < 8; ++r)
      out[((size_t)b * L_ + q0 + r + 8 * half) * D_ + col] = o[t][r];
  }
}

extern "C" void kernel_launch(void* const* d_in, const int* in_sizes, int n_in,
                              void* d_out, int out_size, void* d_ws, size_t ws_size,
                              hipStream_t stream) {
  (void)in_sizes; (void)n_in; (void)out_size; (void)ws_size;
  const float* x    = (const float*)d_in[0];
  const float* W    = (const float*)d_in[1];
  const float* bias = (const float*)d_in[2];
  float* out = (float*)d_out;

  char* ws = (char*)d_ws;
  size_t off = 0;
  auto alloc = [&](size_t bytes) -> void* {
    void* p = ws + off;
    off += (bytes + 255) & ~(size_t)255;
    return p;
  };
  __bf16* xb    = (__bf16*)alloc((size_t)B_ * L_ * D_ * 2);  // x in bf16 (row major)
  __bf16* xt    = (__bf16*)alloc((size_t)B_ * D_ * L_ * 2);  // x transposed bf16
  __bf16* wb    = (__bf16*)alloc((size_t)D_ * D_ * 2);       // W bf16
  __bf16* xproj = (__bf16*)alloc((size_t)B_ * L_ * D_ * 2);  // relu(xW^T+b) bf16
  __bf16* alpha = (__bf16*)alloc((size_t)B_ * L_ * L_ * 2);  // softmax weights bf16

  sam_cvt_bf16_kernel<<<2048, 256, 0, stream>>>(x, xb, B_ * L_ * D_);
  sam_cvt_bf16_kernel<<<512, 256, 0, stream>>>(W, wb, D_ * D_);
  sam_transpose_kernel<<<dim3(D_ / 16, L_ / 16, B_), dim3(16, 16), 0, stream>>>(x, xt);
  sam_proj_kernel<<<dim3(D_ / 64, (B_ * L_) / 16), 32, 0, stream>>>(xb, wb, bias, xproj);
  sam_softmax_kernel<<<B_ * (L_ / 16), 32, 0, stream>>>(xproj, alpha);
  sam_pv_kernel<<<dim3(D_ / 128, L_ / 16, B_), 32, 0, stream>>>(alpha, xt, out);
}
